// MemoryBank_919123002042
// MI455X (gfx1250) — matrix operand order repaired and checked by reading, not compiled, not run on hardware
//
#include <hip/hip_runtime.h>
#include <math.h>

// MemoryBank retrieval for MI455X (gfx1250), wave32 + WMMA f16->f32.
// v4: 32 key rows per iteration, two independent WMMA accumulator chains,
// software-pipelined A-fragment loads from double-buffered LDS (so WMMAs wait
// on partial dscnt, not dscnt==0), key tiles + index_map shared by 8 waves.

#define DDIM 256
#define TOPK 5
#define NCHUNKS 125
#define TEMP_INV 10.0f
#define EPSN 1e-12f

typedef _Float16 h8   __attribute__((ext_vector_type(8)));
typedef _Float16 v16h __attribute__((ext_vector_type(16)));
typedef float    v8f  __attribute__((ext_vector_type(8)));
typedef float    f4   __attribute__((ext_vector_type(4)));

struct P { float v; int i; };
static __device__ __forceinline__ P pmax(P a, P b) { return (a.v >= b.v) ? a : b; }
static __device__ __forceinline__ P pmin(P a, P b) { return (a.v <= b.v) ? a : b; }
static __device__ __forceinline__ int imin(int a, int b) { return a < b ? a : b; }
static __device__ __forceinline__ int imax(int a, int b) { return a > b ? a : b; }

// Bubble-insert candidate into sorted-descending 5-list (branchless, reg-only).
static __device__ __forceinline__ void insert5(P t[TOPK], P c) {
#pragma unroll
  for (int j = 0; j < TOPK; ++j) {
    P hi = pmax(t[j], c);
    c    = pmin(t[j], c);
    t[j] = hi;
  }
}

// Merge two sorted-descending 5-lists into top-5 of union.
// kth largest (0-idx k) = max over splits: a[k], b[k], min(a[i-1], b[k-i]).
static __device__ __forceinline__ void merge5(const P a[TOPK], const P b[TOPK], P r[TOPK]) {
#pragma unroll
  for (int k = 0; k < TOPK; ++k) {
    P best = pmax(a[k], b[k]);
#pragma unroll
    for (int i = 1; i <= k; ++i)
      best = pmax(best, pmin(a[i - 1], b[k - i]));
    r[k] = best;
  }
}

// --- Kernel 1: L2-normalize rows (fp32 -> f16), optionally emit 1/max(||x||,eps).
__global__ void mb_normalize_rows(const float* __restrict__ x,
                                  _Float16* __restrict__ y,
                                  float* __restrict__ invn, int rows) {
  int gtid = blockIdx.x * blockDim.x + threadIdx.x;
  int row  = gtid >> 5;
  int lane = threadIdx.x & 31;
  if (row >= rows) return;
  const float* r = x + (size_t)row * DDIM + lane * 8;
  f4 a = *(const f4*)(r);
  f4 b = *(const f4*)(r + 4);
  float s = a.x * a.x + a.y * a.y + a.z * a.z + a.w * a.w +
            b.x * b.x + b.y * b.y + b.z * b.z + b.w * b.w;
#pragma unroll
  for (int m = 16; m >= 1; m >>= 1) s += __shfl_xor(s, m, 32);
  float sc = 1.0f / fmaxf(sqrtf(s), EPSN);
  h8 o;
  o[0] = (_Float16)(a.x * sc); o[1] = (_Float16)(a.y * sc);
  o[2] = (_Float16)(a.z * sc); o[3] = (_Float16)(a.w * sc);
  o[4] = (_Float16)(b.x * sc); o[5] = (_Float16)(b.y * sc);
  o[6] = (_Float16)(b.z * sc); o[7] = (_Float16)(b.w * sc);
  *(h8*)(y + (size_t)row * DDIM + lane * 8) = o;
  if (invn && lane == 0) invn[row] = sc;
}

// --- Kernel 2: fused WMMA GEMM (keys=M, queries=N) + exclusion + per-chunk top-5.
// Block = 8 waves sharing one key chunk (blockIdx.y); wave w handles query tile
// blockIdx.x*8+w. Each iteration covers a PAIR of 16-row key tiles (32 rows,
// 16 KB f16) double-buffered in LDS with +16B row padding; index_map values
// staged alongside. Two WMMA chains/iter, A-frags software-pipelined.
#define ROWU4 33  // uint4 per LDS row (32 data + 1 pad)
__global__ void __launch_bounds__(256) mb_sim_topk(
    const _Float16* __restrict__ kn, const _Float16* __restrict__ qn,
    const int* __restrict__ imap, const int* __restrict__ excl,
    float* __restrict__ pvals, int* __restrict__ pidx,
    int N, int Bq, int tilesPerChunk, int totalTiles) {
  __shared__ uint4 sm[2][32 * ROWU4];
  __shared__ int   simap[2][32];

  int lane   = threadIdx.x & 31;
  int wib    = threadIdx.x >> 5;
  int chunk  = blockIdx.y;                      // uniform per block
  int qtiles = (Bq + 15) >> 4;
  int qt     = imin(blockIdx.x * 8 + wib, qtiles - 1);

  int hi  = lane >> 4;   // half-lane selector
  int col = lane & 15;   // query column / A-matrix key row (mod 16)
  int q   = imin(qt * 16 + col, Bq - 1);

  // Loop-invariant B fragments (queries), contiguous k order, same as A:
  // lane<16 -> K 0..15, lane>=16 -> K 16..31 of each 32-dim chunk.
  v16h bf[8];
  const _Float16* qbase = qn + (size_t)q * DDIM;
#pragma unroll
  for (int kk = 0; kk < 8; ++kk)
    bf[kk] = *(const v16h*)(qbase + kk * 32 + hi * 16);

  int ex = excl[q];

  P t[TOPK];
#pragma unroll
  for (int j = 0; j < TOPK; ++j) { t[j].v = -__builtin_inff(); t[j].i = -1; }

  int kt0 = chunk * tilesPerChunk;
  int kt1 = imin(kt0 + tilesPerChunk, totalTiles);

  // Staging role: thread copies one 64B chunk; row = tid/8 (0..31),
  // 64B column = tid%8. Plus threads 0..31 stage index_map for the 32 rows.
  int srow = threadIdx.x >> 3;
  int scol = threadIdx.x & 7;

  if (kt0 < kt1) {  // stage first pair-tile into buffer 0
    int gr = imin(kt0 * 16 + srow, N - 1);
    const uint4* src = (const uint4*)(kn + (size_t)gr * DDIM) + scol * 4;
    uint4* dst = &sm[0][srow * ROWU4 + scol * 4];
    dst[0] = src[0]; dst[1] = src[1]; dst[2] = src[2]; dst[3] = src[3];
    if (threadIdx.x < 32)
      simap[0][threadIdx.x] = imap[imin(kt0 * 16 + (int)threadIdx.x, N - 1)];
  }
  __syncthreads();

  for (int kt = kt0; kt < kt1; kt += 2) {
    int  cur     = ((kt - kt0) >> 1) & 1;
    bool hasNext = (kt + 2 < kt1);              // block-uniform
    bool tile2ok = (kt + 1 < kt1);              // block-uniform
    uint4 n0, n1, n2, n3;
    int   nid = 0;
    if (hasNext) {  // issue next pair-tile's global loads before compute
      int gr = imin((kt + 2) * 16 + srow, N - 1);
      const uint4* src = (const uint4*)(kn + (size_t)gr * DDIM) + scol * 4;
      n0 = src[0]; n1 = src[1]; n2 = src[2]; n3 = src[3];
      if (threadIdx.x < 32)
        nid = imap[imin((kt + 2) * 16 + (int)threadIdx.x, N - 1)];
    }

    // Two independent accumulator chains; A-fragment loads pipelined one
    // kk-step ahead so each WMMA pair waits on a partial dscnt only.
    const uint4* rb0 = &sm[cur][col * ROWU4];
    const uint4* rb1 = &sm[cur][(16 + col) * ROWU4];
    v8f acc0 = {}, acc1 = {};

    h8 c0lo = *(const h8*)(rb0 + 2 * hi);
    h8 c0hi = *(const h8*)(rb0 + 2 * hi + 1);
    h8 c1lo = *(const h8*)(rb1 + 2 * hi);
    h8 c1hi = *(const h8*)(rb1 + 2 * hi + 1);
#pragma unroll
    for (int kk = 0; kk < 8; ++kk) {
      h8 p0lo, p0hi, p1lo, p1hi;
      if (kk < 7) {  // prefetch next kk-step's fragments from LDS
        p0lo = *(const h8*)(rb0 + (kk + 1) * 4 + 2 * hi);
        p0hi = *(const h8*)(rb0 + (kk + 1) * 4 + 2 * hi + 1);
        p1lo = *(const h8*)(rb1 + (kk + 1) * 4 + 2 * hi);
        p1hi = *(const h8*)(rb1 + (kk + 1) * 4 + 2 * hi + 1);
      }
      v16h af0 = __builtin_shufflevector(c0lo, c0hi, 0, 1, 2, 3, 4, 5, 6, 7,
                                         8, 9, 10, 11, 12, 13, 14, 15);
      v16h af1 = __builtin_shufflevector(c1lo, c1hi, 0, 1, 2, 3, 4, 5, 6, 7,
                                         8, 9, 10, 11, 12, 13, 14, 15);
      acc0 = __builtin_amdgcn_wmma_f32_16x16x32_f16(
          false, af0, false, bf[kk], (short)0, acc0, false, false);
      acc1 = __builtin_amdgcn_wmma_f32_16x16x32_f16(
          false, af1, false, bf[kk], (short)0, acc1, false, false);
      if (kk < 7) {
        c0lo = p0lo; c0hi = p0hi; c1lo = p1lo; c1hi = p1hi;
      }
    }

    // Candidate processing, tile 0 then tile 1 (ids from LDS-staged imap).
#pragma unroll
    for (int tile = 0; tile < 2; ++tile) {
      int  base = (kt + tile) * 16 + hi * 8;
      bool ok   = (tile == 0) || tile2ok;
      const int4* ip = (const int4*)&simap[cur][tile * 16 + hi * 8];
      int4 i0 = ip[0], i1 = ip[1];
      int ids[8] = {i0.x, i0.y, i0.z, i0.w, i1.x, i1.y, i1.z, i1.w};
      float vals[8];
#pragma unroll
      for (int v = 0; v < 8; ++v) {
        float sv = (tile == 0) ? acc0[v] : acc1[v];
        vals[v]  = (!ok || ids[v] == ex || base + v >= N) ? -__builtin_inff() : sv;
      }
      float vmax = vals[0];
#pragma unroll
      for (int v = 1; v < 8; ++v) vmax = fmaxf(vmax, vals[v]);
      if (vmax > t[4].v) {
#pragma unroll
        for (int v = 0; v < 8; ++v) {
          if (vals[v] > t[4].v) {
            P c; c.v = vals[v]; c.i = base + v;
            insert5(t, c);
          }
        }
      }
    }

    if (hasNext) {  // commit next pair-tile to the other buffer
      uint4* dst = &sm[1 - cur][srow * ROWU4 + scol * 4];
      dst[0] = n0; dst[1] = n1; dst[2] = n2; dst[3] = n3;
      if (threadIdx.x < 32) simap[1 - cur][threadIdx.x] = nid;
    }
    __syncthreads();  // next buffer ready; current buffer free for overwrite
  }

  // Merge the two half-lane lists covering the same query column.
  P o[TOPK];
#pragma unroll
  for (int j = 0; j < TOPK; ++j) {
    o[j].v = __shfl_xor(t[j].v, 16, 32);
    o[j].i = __shfl_xor(t[j].i, 16, 32);
  }
  P r[TOPK];
  merge5(t, o, r);
  if (hi == 0 && (qt * 16 + col) < Bq && (blockIdx.x * 8 + wib) < qtiles) {
    size_t base = ((size_t)q * NCHUNKS + chunk) * TOPK;
#pragma unroll
    for (int j = 0; j < TOPK; ++j) { pvals[base + j] = r[j].v; pidx[base + j] = r[j].i; }
  }
}

// --- Kernel 3: merge 125 partial top-5 lists per query, softmax, gather,
// renormalize. One wave per query.
__global__ void mb_finalize(const float* __restrict__ pvals,
                            const int* __restrict__ pidx,
                            const float* __restrict__ keys,
                            const float* __restrict__ invn,
                            float* __restrict__ out, int Bq, int N) {
  int lane = threadIdx.x & 31;
  int q    = (blockIdx.x * blockDim.x + threadIdx.x) >> 5;
  if (q >= Bq) return;

  P t[TOPK];
#pragma unroll
  for (int j = 0; j < TOPK; ++j) { t[j].v = -__builtin_inff(); t[j].i = -1; }
  size_t base = (size_t)q * NCHUNKS * TOPK;
  for (int j = lane; j < NCHUNKS * TOPK; j += 32) {
    P c; c.v = pvals[base + j]; c.i = pidx[base + j];
    if (c.v > t[4].v) insert5(t, c);
  }
  // Butterfly merge: all lanes converge to the global sorted top-5.
#pragma unroll
  for (int m = 16; m >= 1; m >>= 1) {
    P o[TOPK];
#pragma unroll
    for (int j = 0; j < TOPK; ++j) {
      o[j].v = __shfl_xor(t[j].v, m, 32);
      o[j].i = __shfl_xor(t[j].i, m, 32);
    }
    P r[TOPK];
    merge5(t, o, r);
#pragma unroll
    for (int j = 0; j < TOPK; ++j) t[j] = r[j];
  }

  // softmax(top_sim / 0.1) with max subtraction (t[0] is the max)
  float e[TOPK], s = 0.0f;
#pragma unroll
  for (int k = 0; k < TOPK; ++k) { e[k] = __expf((t[k].v - t[0].v) * TEMP_INV); s += e[k]; }
  float w[TOPK];
#pragma unroll
  for (int k = 0; k < TOPK; ++k) w[k] = e[k] / s;

  // Weighted gather of fp32-normalized key rows; each lane owns 8 dims.
  int d0 = lane * 8;
  float acc[8] = {0, 0, 0, 0, 0, 0, 0, 0};
#pragma unroll
  for (int k = 0; k < TOPK; ++k) {
    int   idx = t[k].i < 0 ? 0 : t[k].i;
    float sc  = (t[k].i < 0 ? 0.0f : w[k]) * invn[idx];
    const float* kr = keys + (size_t)idx * DDIM + d0;
    f4 a = *(const f4*)kr;
    f4 b = *(const f4*)(kr + 4);
    acc[0] += sc * a.x; acc[1] += sc * a.y; acc[2] += sc * a.z; acc[3] += sc * a.w;
    acc[4] += sc * b.x; acc[5] += sc * b.y; acc[6] += sc * b.z; acc[7] += sc * b.w;
  }
  float ss = 0.0f;
#pragma unroll
  for (int i = 0; i < 8; ++i) ss += acc[i] * acc[i];
#pragma unroll
  for (int m = 16; m >= 1; m >>= 1) ss += __shfl_xor(ss, m, 32);
  float sc2 = 1.0f / fmaxf(sqrtf(ss), EPSN);

  float* od = out + (size_t)q * DDIM + d0;
  f4 oa = {acc[0] * sc2, acc[1] * sc2, acc[2] * sc2, acc[3] * sc2};
  f4 ob = {acc[4] * sc2, acc[5] * sc2, acc[6] * sc2, acc[7] * sc2};
  *(f4*)od       = oa;
  *(f4*)(od + 4) = ob;

  if (lane == 0) {
    out[(size_t)Bq * DDIM + q] = 1.0f - t[0].v;  // distances
    float* wout = out + (size_t)Bq * DDIM + Bq + (size_t)q * TOPK;
#pragma unroll
    for (int k = 0; k < TOPK; ++k) wout[k] = w[k];
  }
}

extern "C" void kernel_launch(void* const* d_in, const int* in_sizes, int n_in,
                              void* d_out, int out_size, void* d_ws, size_t ws_size,
                              hipStream_t stream) {
  const float* query = (const float*)d_in[0];
  const float* keys  = (const float*)d_in[1];
  const int*   imap  = (const int*)d_in[2];
  const int*   excl  = (const int*)d_in[3];
  int N = in_sizes[2];
  int B = in_sizes[3];

  // Workspace carve (~62 MB): kn f16, qn f16, invn, partial top-5 vals/idx.
  char* ws = (char*)d_ws;
  size_t off = 0;
  auto carve = [&](size_t bytes) -> void* {
    void* p = ws + off;
    off = (off + bytes + 255) & ~(size_t)255;
    return p;
  };
  _Float16* kn   = (_Float16*)carve((size_t)N * DDIM * sizeof(_Float16));
  _Float16* qn   = (_Float16*)carve((size_t)B * DDIM * sizeof(_Float16));
  float*    invn = (float*)carve((size_t)N * sizeof(float));
  float*    pv   = (float*)carve((size_t)B * NCHUNKS * TOPK * sizeof(float));
  int*      pi   = (int*)carve((size_t)B * NCHUNKS * TOPK * sizeof(int));

  int totalTiles = (N + 15) / 16;                        // 6250
  int tpc        = (totalTiles + NCHUNKS - 1) / NCHUNKS; // 50
  int qtiles     = (B + 15) / 16;                        // 128

  mb_normalize_rows<<<dim3((N + 7) / 8), dim3(256), 0, stream>>>(keys, kn, invn, N);
  mb_normalize_rows<<<dim3((B + 7) / 8), dim3(256), 0, stream>>>(query, qn, nullptr, B);
  mb_sim_topk<<<dim3((qtiles + 7) / 8, NCHUNKS), dim3(256), 0, stream>>>(
      kn, qn, imap, excl, pv, pi, N, B, tpc, totalTiles);
  mb_finalize<<<dim3((B + 7) / 8), dim3(256), 0, stream>>>(pv, pi, keys, invn,
                                                           (float*)d_out, B, N);
}
